// MoE_79714593014220
// MI455X (gfx1250) — matrix operand order repaired and checked by reading, not compiled, not run on hardware
//
#include <hip/hip_runtime.h>

typedef __bf16 bf16;
typedef bf16  v16bf __attribute__((ext_vector_type(16)));
typedef float v8f   __attribute__((ext_vector_type(8)));

#define N_TOK 2048
#define DIM   1024
#define FFD   4096
#define NEXP  8
#define NKROW 4096   // total assignments = 2 * N_TOK (top-2)

// ---------------------------------------------------------------- router ----
__global__ void __launch_bounds__(256)
router_k(const float* __restrict__ x, const float* __restrict__ gw,
         int* __restrict__ cnt, float* __restrict__ imp,
         int* __restrict__ bucket_t, int4* __restrict__ tokmeta,
         float2* __restrict__ tokgate, float* __restrict__ muv,
         float* __restrict__ rsv) {
  const int t    = blockIdx.x;
  const int tid  = threadIdx.x;
  const int w    = tid >> 5;
  const int lane = tid & 31;
  __shared__ float slog[NEXP];
  __shared__ float ssum[NEXP];
  __shared__ float ssq[NEXP];

  const float* xr = x  + (long)t * DIM;
  const float* gr = gw + (long)w * DIM;
  float acc = 0.f, s = 0.f, s2 = 0.f;
#pragma unroll
  for (int i = 0; i < DIM / 128; ++i) {
    int k = (i * 32 + lane) * 4;
    float4 xv = *(const float4*)(xr + k);
    float4 gv = *(const float4*)(gr + k);
    acc += xv.x * gv.x + xv.y * gv.y + xv.z * gv.z + xv.w * gv.w;
    s   += xv.x + xv.y + xv.z + xv.w;
    s2  += xv.x * xv.x + xv.y * xv.y + xv.z * xv.z + xv.w * xv.w;
  }
#pragma unroll
  for (int o = 16; o > 0; o >>= 1) {
    acc += __shfl_down(acc, o, 32);
    s   += __shfl_down(s,   o, 32);
    s2  += __shfl_down(s2,  o, 32);
  }
  if (lane == 0) { slog[w] = acc; ssum[w] = s; ssq[w] = s2; }
  __syncthreads();

  if (tid == 0) {
    const float mu  = ssum[0] * (1.0f / DIM);
    const float var = ssq[0] * (1.0f / DIM) - mu * mu;
    const float rs  = rsqrtf(var + 1e-6f);
    muv[t] = mu;
    rsv[t] = rs;

    float lg[NEXP];
    float mx = -1e30f;
#pragma unroll
    for (int e = 0; e < NEXP; ++e) { lg[e] = slog[e]; mx = fmaxf(mx, lg[e]); }
    float se = 0.f;
#pragma unroll
    for (int e = 0; e < NEXP; ++e) { lg[e] = __expf(lg[e] - mx); se += lg[e]; }
    const float inv = 1.0f / se;
#pragma unroll
    for (int e = 0; e < NEXP; ++e) { lg[e] *= inv; atomicAdd(&imp[e], lg[e]); }

    int i0 = 0;
#pragma unroll
    for (int e = 1; e < NEXP; ++e) if (lg[e] > lg[i0]) i0 = e;
    int i1 = (i0 == 0) ? 1 : 0;
#pragma unroll
    for (int e = 0; e < NEXP; ++e) if (e != i0 && lg[e] > lg[i1]) i1 = e;

    const int s0 = atomicAdd(&cnt[i0], 1);
    const int s1 = atomicAdd(&cnt[i1], 1);
    bucket_t[i0 * N_TOK + s0] = t;
    bucket_t[i1 * N_TOK + s1] = t;
    tokmeta[t] = make_int4(i0, s0, i1, s1);
    tokgate[t] = make_float2(lg[i0], lg[i1]);
  }
}

// -------------------------------------------------- offsets + aux loss -----
__global__ void finalize_k(const int* __restrict__ cnt, const float* __restrict__ imp,
                           int* __restrict__ off, float* __restrict__ aux_out) {
  if (threadIdx.x == 0 && blockIdx.x == 0) {
    int o = 0;
    float a = 0.f;
#pragma unroll
    for (int e = 0; e < NEXP; ++e) {
      off[e] = o;
      o += cnt[e];
      a += imp[e] * (float)cnt[e];
    }
    *aux_out = a * (float)NEXP / ((float)N_TOK * (float)N_TOK);
  }
}

// ------------------------------------- gather + LN affine -> bf16 A rows ---
__global__ void __launch_bounds__(256)
gather_k(const float* __restrict__ x, const float* __restrict__ lng,
         const float* __restrict__ lnb, const int* __restrict__ cnt,
         const int* __restrict__ off, const int* __restrict__ bucket_t,
         const float* __restrict__ muv, const float* __restrict__ rsv,
         bf16* __restrict__ Xe) {
  const int e    = blockIdx.y;
  const int slot = blockIdx.x;
  if (slot >= cnt[e]) return;
  const int  t   = bucket_t[e * N_TOK + slot];
  const long row = (long)off[e] + slot;
  const float mu = muv[t], rs = rsv[t];
  const int k = threadIdx.x * 4;                     // 256 threads * 4 = 1024
  const float4 xv = *(const float4*)(x   + (long)t * DIM + k);
  const float4 gv = *(const float4*)(lng + (long)e * DIM + k);
  const float4 bv = *(const float4*)(lnb + (long)e * DIM + k);
  union { bf16 h[4]; uint2 u; } o;
  o.h[0] = (bf16)((xv.x - mu) * rs * gv.x + bv.x);
  o.h[1] = (bf16)((xv.y - mu) * rs * gv.y + bv.y);
  o.h[2] = (bf16)((xv.z - mu) * rs * gv.z + bv.z);
  o.h[3] = (bf16)((xv.w - mu) * rs * gv.w + bv.w);
  *(uint2*)(Xe + row * DIM + k) = o.u;
}

// ------------------------------------------------------- WMMA tile loads ---
// A 16x32 bf16: lane(half,idx): elems 0..15 <-> K = half*16+0..15
// -> two contiguous 16-byte chunks.
__device__ __forceinline__ v16bf load_a_bf16(const bf16* row, int ka) {
  union { v16bf v; uint4 u[2]; } a;
  a.u[0] = *(const uint4*)(row + ka);
  a.u[1] = *(const uint4*)(row + ka + 8);
  return a.v;
}
// B 32x16 bf16, same k order as A: lane(half,n): elems 0..15 <-> K = half*16+0..15
// -> 16 contiguous fp32 from the weight row, packed to bf16.
__device__ __forceinline__ v16bf load_b_f32(const float* p) {
  float4 f[4];
  f[0] = ((const float4*)p)[0];
  f[1] = ((const float4*)p)[1];
  f[2] = ((const float4*)p)[2];
  f[3] = ((const float4*)p)[3];
  const float* s = (const float*)f;
  v16bf b;
#pragma unroll
  for (int i = 0; i < 16; ++i) b[i] = (bf16)s[i];
  return b;
}

// --------------------------------- FFN1: H = relu(Xe @ w1^T + b1) (bf16) ---
__global__ void __launch_bounds__(256)
ffn1_k(const bf16* __restrict__ Xe, const float* __restrict__ w1,
       const float* __restrict__ b1, const int* __restrict__ cnt,
       const int* __restrict__ off, bf16* __restrict__ H) {
  const int e       = blockIdx.z;
  const int rowend  = off[e] + cnt[e];
  const int rowbase = off[e] + blockIdx.y * 64;
  if (rowbase >= rowend) return;
  const int fbase = blockIdx.x * 64;

  const int w    = threadIdx.x >> 5;
  const int lane = threadIdx.x & 31;
  const int mi   = w >> 2;          // 0..1  (rows mi*32 .. +31)
  const int fi   = w & 3;           // 0..3  (cols fi*16 .. +15)
  const int half = lane >> 4;
  const int idx  = lane & 15;

  const int f     = fbase + fi * 16 + idx;
  const int rlast = rowend - 1;
  const int r0    = rowbase + mi * 32 + idx;
  const int r1    = r0 + 16;
  // clamp to a valid row: garbage rows only affect D rows we never store
  const bf16*  a0p = Xe + (long)min(r0, rlast) * DIM;
  const bf16*  a1p = Xe + (long)min(r1, rlast) * DIM;
  const float* bp  = w1 + ((long)e * FFD + f) * DIM;

  v8f c0 = {};
  v8f c1 = {};
#pragma unroll 2
  for (int kk = 0; kk < DIM; kk += 32) {
    const int ka = kk + half * 16;
    v16bf a0 = load_a_bf16(a0p, ka);
    v16bf a1 = load_a_bf16(a1p, ka);
    v16bf b  = load_b_f32(bp + kk + half * 16);
    c0 = __builtin_amdgcn_wmma_f32_16x16x32_bf16(false, a0, false, b, (short)0, c0, false, false);
    c1 = __builtin_amdgcn_wmma_f32_16x16x32_bf16(false, a1, false, b, (short)0, c1, false, false);
  }

  const float bias = b1[(long)e * FFD + f];
#pragma unroll
  for (int r = 0; r < 8; ++r) {
    const int m0 = rowbase + mi * 32 + half * 8 + r;
    if (m0 < rowend) H[(long)m0 * FFD + f] = (bf16)fmaxf(c0[r] + bias, 0.f);
    const int m1 = m0 + 16;
    if (m1 < rowend) H[(long)m1 * FFD + f] = (bf16)fmaxf(c1[r] + bias, 0.f);
  }
}

// ----------------------------------- FFN2: Y = H @ w2^T + b2  (fp32 out) ---
__global__ void __launch_bounds__(256)
ffn2_k(const bf16* __restrict__ H, const float* __restrict__ w2,
       const float* __restrict__ b2, const int* __restrict__ cnt,
       const int* __restrict__ off, float* __restrict__ Y) {
  const int e       = blockIdx.z;
  const int rowend  = off[e] + cnt[e];
  const int rowbase = off[e] + blockIdx.y * 64;
  if (rowbase >= rowend) return;
  const int dbase = blockIdx.x * 64;

  const int w    = threadIdx.x >> 5;
  const int lane = threadIdx.x & 31;
  const int mi   = w >> 2;
  const int fi   = w & 3;
  const int half = lane >> 4;
  const int idx  = lane & 15;

  const int d     = dbase + fi * 16 + idx;
  const int rlast = rowend - 1;
  const int r0    = rowbase + mi * 32 + idx;
  const int r1    = r0 + 16;
  const bf16*  a0p = H + (long)min(r0, rlast) * FFD;
  const bf16*  a1p = H + (long)min(r1, rlast) * FFD;
  const float* bp  = w2 + ((long)e * DIM + d) * FFD;

  v8f c0 = {};
  v8f c1 = {};
#pragma unroll 2
  for (int kk = 0; kk < FFD; kk += 32) {
    const int ka = kk + half * 16;
    v16bf a0 = load_a_bf16(a0p, ka);
    v16bf a1 = load_a_bf16(a1p, ka);
    v16bf b  = load_b_f32(bp + kk + half * 16);
    c0 = __builtin_amdgcn_wmma_f32_16x16x32_bf16(false, a0, false, b, (short)0, c0, false, false);
    c1 = __builtin_amdgcn_wmma_f32_16x16x32_bf16(false, a1, false, b, (short)0, c1, false, false);
  }

  const float bias = b2[(long)e * DIM + d];
#pragma unroll
  for (int r = 0; r < 8; ++r) {
    const int m0 = rowbase + mi * 32 + half * 8 + r;
    if (m0 < rowend) Y[(long)m0 * DIM + d] = c0[r] + bias;
    const int m1 = m0 + 16;
    if (m1 < rowend) Y[(long)m1 * DIM + d] = c1[r] + bias;
  }
}

// --------------------- combine: out = g0*Y0 + g1*Y1 + (g0+g1)*x (+skip) ----
__global__ void __launch_bounds__(256)
combine_k(const float* __restrict__ x, const float* __restrict__ Y,
          const int* __restrict__ off, const int4* __restrict__ tokmeta,
          const float2* __restrict__ tokgate, float* __restrict__ out) {
  const int t = blockIdx.x;
  const int4   m = tokmeta[t];
  const float2 g = tokgate[t];
  const long r0 = (long)off[m.x] + m.y;
  const long r1 = (long)off[m.z] + m.w;
  const int  k  = threadIdx.x * 4;                   // 256 threads * 4 = 1024
  const float4 y0 = *(const float4*)(Y + r0 * DIM + k);
  const float4 y1 = *(const float4*)(Y + r1 * DIM + k);
  const float4 xv = *(const float4*)(x + (long)t * DIM + k);
  const float gs = g.x + g.y;
  float4 o;
  o.x = g.x * y0.x + g.y * y1.x + gs * xv.x;
  o.y = g.x * y0.y + g.y * y1.y + gs * xv.y;
  o.z = g.x * y0.z + g.y * y1.z + gs * xv.z;
  o.w = g.x * y0.w + g.y * y1.w + gs * xv.w;
  *(float4*)(out + (long)t * DIM + k) = o;
}

// ---------------------------------------------------------------- launch ---
extern "C" void kernel_launch(void* const* d_in, const int* in_sizes, int n_in,
                              void* d_out, int out_size, void* d_ws, size_t ws_size,
                              hipStream_t stream) {
  const float* x    = (const float*)d_in[0];
  const float* gw   = (const float*)d_in[1];
  const float* lng  = (const float*)d_in[2];
  const float* lnb  = (const float*)d_in[3];
  const float* w1   = (const float*)d_in[4];
  const float* b1   = (const float*)d_in[5];
  const float* w2   = (const float*)d_in[6];
  const float* b2   = (const float*)d_in[7];
  float* out = (float*)d_out;
  char*  ws  = (char*)d_ws;

  // workspace layout (bytes)
  int*    cnt      = (int*)   (ws + 0);
  float*  imp      = (float*) (ws + 256);
  int*    off      = (int*)   (ws + 512);
  int*    bucket_t = (int*)   (ws + 4096);                 // 64 KB
  int4*   tokmeta  = (int4*)  (ws + 4096 + 65536);         // 32 KB
  float2* tokgate  = (float2*)(ws + 4096 + 65536 + 32768); // 16 KB
  float*  muv      = (float*) (ws + 4096 + 65536 + 32768 + 16384);
  float*  rsv      = (float*) (ws + 4096 + 65536 + 32768 + 16384 + 8192);
  bf16*   Xe       = (bf16*)  (ws + (1L << 20));           //  8 MB
  bf16*   H        = (bf16*)  (ws + (16L << 20));          // 32 MB
  float*  Y        = (float*) (ws + (48L << 20));          // 16 MB  (ends at 64 MB)

  // zero the atomic counters / importance sums each call
  hipMemsetAsync(ws, 0, 1024, stream);

  router_k<<<N_TOK, 256, 0, stream>>>(x, gw, cnt, imp, bucket_t, tokmeta, tokgate, muv, rsv);
  finalize_k<<<1, 32, 0, stream>>>(cnt, imp, off, out + (long)N_TOK * DIM);
  gather_k<<<dim3(N_TOK, NEXP), 256, 0, stream>>>(x, lng, lnb, cnt, off, bucket_t, muv, rsv, Xe);
  ffn1_k<<<dim3(FFD / 64, N_TOK / 64, NEXP), 256, 0, stream>>>(Xe, w1, b1, cnt, off, H);
  ffn2_k<<<dim3(DIM / 64, N_TOK / 64, NEXP), 256, 0, stream>>>(H, w2, b2, cnt, off, Y);
  combine_k<<<N_TOK, 256, 0, stream>>>(x, Y, off, tokmeta, tokgate, out);
}
